// Qwen3NextGatedDeltaNetPrefill_28750511079922
// MI455X (gfx1250) — matrix operand order repaired and checked by reading, not compiled, not run on hardware
//
#include <hip/hip_runtime.h>

// Qwen3-Next GatedDeltaNet prefill for MI455X (gfx1250), chunked delta-rule with WMMA.
// B=2, S=2048, HK=16, HV=32, DK=DV=128, conv K=4, chunk L=64.
// All WMMA A/B fragments are laid out so every lane's 16-half fragment is two
// contiguous 16B runs -> ds_load_b128, with padded strides for bank spreading.

#define BB 2
#define SS 2048
#define HKH 16
#define HVH 32
#define DKD 128
#define DVD 128
#define KW 4
#define QKVC 8192
#define LCH 64
#define NCHUNK (SS / LCH)
#define EPSF 1e-6f
#define SCALEF 0.08838834764831845f   // 128^-0.5

#define SSTR 136   // S16T row stride (halves): 16B-aligned, bank-spread
#define USTR 72    // uT   row stride (halves)
#define K2STR 72   // k2T  row stride (halves)

typedef __attribute__((ext_vector_type(16))) _Float16 v16h;
typedef __attribute__((ext_vector_type(8)))  float    v8f;

static __device__ __forceinline__ v8f wmma16(v16h a, v16h b, v8f c) {
  // D = A(16x32 f16) * B(32x16 f16) + C(16x16 f32)
  return __builtin_amdgcn_wmma_f32_16x16x32_f16(false, a, false, b, (short)0, c, false, false);
}

// ---- wave32 WMMA fragment loaders (layouts per CDNA5 ISA 7.12.2) ----
// A fragment: lane l row m=r0+(l&15); element j -> K = c0 + 8*(l>=16) + j + (j>=8 ? 8 : 0)
// With cs==1 each lane reads two contiguous 16B runs -> ds_load_b128.
static __device__ __forceinline__ v16h frag_a(const _Float16* p, int r0, int c0, int rs, int cs) {
  int lane = threadIdx.x & 31;
  int m = r0 + (lane & 15);
  int kb = (lane >> 4) * 8;
  v16h a;
#pragma unroll
  for (int j = 0; j < 16; ++j) {
    int k = c0 + kb + j + ((j >= 8) ? 8 : 0);
    a[j] = p[m * rs + k * cs];
  }
  return a;
}

// B fragment: lane l col n=c0+(l&15); element j -> K = r0 + 8*(l>=16) + j + (j>=8 ? 8 : 0)
// Expects column-major storage p[n][k] (rs==1): two contiguous 16B runs per lane.
static __device__ __forceinline__ v16h frag_b(const _Float16* p, int r0, int c0, int rs, int cs) {
  int lane = threadIdx.x & 31;
  int n = c0 + (lane & 15);
  int kb = (lane >> 4) * 8;
  v16h b;
#pragma unroll
  for (int j = 0; j < 16; ++j) {
    int k = r0 + kb + j + ((j >= 8) ? 8 : 0);
    b[j] = p[k * rs + n * cs];
  }
  return b;
}

// C/D fragment: lane l col n=c0+(l&15); element i -> row = r0 + 8*(l>=16) + i
static __device__ __forceinline__ v8f load_c(const float* p, int r0, int c0, int rs) {
  int lane = threadIdx.x & 31;
  int n = c0 + (lane & 15);
  int mb = (lane >> 4) * 8;
  v8f c;
#pragma unroll
  for (int i = 0; i < 8; ++i) c[i] = p[(r0 + mb + i) * rs + n];
  return c;
}

static __device__ __forceinline__ void store_c(float* p, int r0, int c0, int rs, v8f c) {
  int lane = threadIdx.x & 31;
  int n = c0 + (lane & 15);
  int mb = (lane >> 4) * 8;
#pragma unroll
  for (int i = 0; i < 8; ++i) p[(r0 + mb + i) * rs + n] = c[i];
}

// ======================= Kernel 1: conv + SiLU + l2norm + gating ========================
__global__ void __launch_bounds__(256) gdn_prep(
    const float* __restrict__ x, const float* __restrict__ bvec,
    const float* __restrict__ avec, const float* __restrict__ cw,
    const float* __restrict__ dtb, const float* __restrict__ alog,
    _Float16* __restrict__ qf, _Float16* __restrict__ kf,
    _Float16* __restrict__ vf, float* __restrict__ gb, float* __restrict__ bbt) {
  __shared__ float y[QKVC];
  __shared__ float rnorm[32];
  int n = blockIdx.x;
  int bidx = n / SS, t = n % SS;
  int tid = threadIdx.x;

  // depthwise causal conv (zero history at sequence start) + SiLU
  for (int c = tid; c < QKVC; c += 256) {
    float acc = 0.f;
#pragma unroll
    for (int j = 0; j < KW; ++j) {
      int tp = t - (KW - 1) + j;
      if (tp >= 0) acc += x[(size_t)(bidx * SS + tp) * QKVC + c] * cw[c * KW + j];
    }
    y[c] = acc / (1.f + __expf(-acc));
  }
  __syncthreads();

  // l2 norms for the 32 q/k heads (first 4096 channels)
  int wave = tid >> 5, lane = tid & 31;
  for (int h = wave; h < 32; h += 8) {
    float s = 0.f;
    for (int i = lane; i < DKD; i += 32) { float v = y[h * DKD + i]; s += v * v; }
#pragma unroll
    for (int off = 16; off > 0; off >>= 1) s += __shfl_xor(s, off, 32);
    if (lane == 0) rnorm[h] = rsqrtf(s + EPSF);
  }
  __syncthreads();

  for (int c = tid; c < QKVC; c += 256) {
    float v = y[c];
    if (c < 2 * HKH * DKD) v *= rnorm[c >> 7];
    _Float16 h = (_Float16)v;
    if (c < HKH * DKD)            qf[(size_t)n * (HKH * DKD) + c] = h;
    else if (c < 2 * HKH * DKD)   kf[(size_t)n * (HKH * DKD) + (c - HKH * DKD)] = h;
    else                          vf[(size_t)n * (HVH * DVD) + (c - 2 * HKH * DKD)] = h;
  }

  if (tid < HVH) {
    int h = tid;
    float aa = avec[(size_t)n * HVH + h] + dtb[h];
    float sp = (aa > 20.f) ? aa : log1pf(__expf(aa));     // softplus
    gb[(size_t)n * HVH + h] = -__expf(alog[h]) * sp;
    float bv = bvec[(size_t)n * HVH + h];
    bbt[(size_t)n * HVH + h] = 1.f / (1.f + __expf(-bv)); // sigmoid
  }
}

// ===================== Kernel 2: chunked gated delta rule (WMMA) ========================
// One workgroup (8 wave32) per (b, hv). fp32 state master + transposed f16 mirror in LDS.
__global__ void __launch_bounds__(256) gdn_chunks(
    const _Float16* __restrict__ qf, const _Float16* __restrict__ kf,
    const _Float16* __restrict__ vf, const float* __restrict__ gb,
    const float* __restrict__ bbt, float* __restrict__ out) {
  extern __shared__ char smem[];
  float*    S32 = (float*)smem;                        // 128x128 f32 state (master)
  float*    Uf  = S32 + DKD * DVD;                     // 64x128 f32: KS0 -> rhs -> U
  float*    Ar  = Uf + LCH * DVD;                      // 64x64 f32: KK^T -> -A (masked)
  float*    QKr = Ar + LCH * LCH;                      // 64x64 f32: QK^T raw
  float*    Gc  = QKr + LCH * LCH;                     // cumulative gate (64)
  float*    eG  = Gc + LCH;                            // exp(G_t)
  float*    w2  = eG + LCH;                            // exp(G_L - G_t)
  float*    bet = w2 + LCH;                            // beta_t
  _Float16* S16T= (_Float16*)(bet + LCH);              // [dv][dk] f16 state, stride SSTR
  _Float16* kc  = S16T + DVD * SSTR;                   // [t][dk]  64x128 f16
  _Float16* qc  = kc + LCH * DKD;                      // [t][dk]  (becomes q*scale*expG)
  _Float16* vc  = qc + LCH * DKD;                      // [t][dv]
  _Float16* k2T = vc + LCH * DVD;                      // [dk][t]  decayed K^T, stride K2STR
  _Float16* uT  = k2T + DKD * K2STR;                   // [dv][s]  f16 U mirror, stride USTR
  _Float16* atth= uT + DVD * USTR;                     // 64x64 masked/decayed attn (row t)
  _Float16* anh = atth + LCH * LCH;                    // 64x64 -A, diag blocks zeroed

  int tid = threadIdx.x, wave = tid >> 5;
  int b = blockIdx.x >> 5, hv = blockIdx.x & 31, hk = hv >> 1;   // GQA repeat 2

  for (int e = tid; e < DKD * DVD; e += 256) S32[e] = 0.f;
  for (int e = tid; e < DVD * SSTR; e += 256) S16T[e] = (_Float16)0.f;
  __syncthreads();

  for (int ch = 0; ch < NCHUNK; ++ch) {
    int tok0 = ch * LCH;
    // ---- load chunk ----
    for (int e = tid; e < LCH * DKD; e += 256) {
      int t = e >> 7, d = e & 127;
      size_t n = (size_t)(b * SS + tok0 + t);
      kc[e] = kf[(n * HKH + hk) * DKD + d];
      qc[e] = qf[(n * HKH + hk) * DKD + d];
      vc[e] = vf[(n * HVH + hv) * DVD + d];
    }
    if (tid < LCH) {
      size_t n = (size_t)(b * SS + tok0 + tid);
      Gc[tid] = gb[n * HVH + hv];
      bet[tid] = bbt[n * HVH + hv];
    }
    __syncthreads();
    if (tid == 0) {  // inclusive prefix of gates (64 elems, trivial)
      float run = 0.f;
      for (int t = 0; t < LCH; ++t) { run += Gc[t]; Gc[t] = run; }
    }
    __syncthreads();
    if (tid < LCH) {
      eG[tid] = __expf(Gc[tid]);
      w2[tid] = __expf(Gc[LCH - 1] - Gc[tid]);
    }
    __syncthreads();

    // ---- stage 1: A_raw = Kc*Kc^T, QK_raw = Qc*Kc^T (32 16x16 tiles) ----
    for (int job = wave; job < 32; job += 8) {
      int prod = job >> 4, tile = job & 15;
      int ti = tile >> 2, tj = tile & 3;
      const _Float16* ap = prod ? qc : kc;
      v8f c = {};
#pragma unroll
      for (int kk = 0; kk < DKD; kk += 32) {
        v16h a = frag_a(ap, ti * 16, kk, DKD, 1);
        v16h bb = frag_b(kc, kk, tj * 16, 1, DKD);   // kc is [n=t][k=dk]: contiguous
        c = wmma16(a, bb, c);
      }
      store_c(prod ? QKr : Ar, ti * 16, tj * 16, LCH, c);
    }
    __syncthreads();

    // ---- stage 2: masks + decay (post-matmul for numerical safety) ----
    for (int e = tid; e < LCH * LCH; e += 256) {
      int t = e >> 6, s = e & 63;
      float d = (s <= t) ? __expf(Gc[t] - Gc[s]) : 0.f;
      atth[e] = (_Float16)((s <= t) ? SCALEF * d * QKr[e] : 0.f);
      float av = (s < t) ? -(bet[t] * d * Ar[e]) : 0.f;  // store NEGATED A
      Ar[e] = av;
      anh[e] = (_Float16)(((s >> 4) != (t >> 4)) ? av : 0.f);  // off-diag blocks only
    }
    for (int e = tid; e < LCH * DKD; e += 256) {       // qd = scale * expG_t * q_t
      int t = e >> 7;
      qc[e] = (_Float16)((float)qc[e] * SCALEF * eG[t]);
    }
    for (int e = tid; e < DKD * LCH; e += 256) {       // k2T[dk][t] = exp(G_L-G_t)*k[t][dk]
      int dk = e >> 6, t = e & 63;
      k2T[dk * K2STR + t] = (_Float16)((float)kc[t * DKD + dk] * w2[t]);
    }
    __syncthreads();

    // ---- stage 3: KS0 = Kc * S0 ----
    for (int job = wave; job < 32; job += 8) {
      int ti = job >> 3, tj = job & 7;
      v8f c = {};
#pragma unroll
      for (int kk = 0; kk < DKD; kk += 32) {
        v16h a = frag_a(kc, ti * 16, kk, DKD, 1);
        v16h bb = frag_b(S16T, kk, tj * 16, 1, SSTR);  // [n=dv][k=dk]: contiguous
        c = wmma16(a, bb, c);
      }
      store_c(Uf, ti * 16, tj * 16, DVD, c);
    }
    __syncthreads();
    // rhs: U = beta*(v - expG * K S0); zero uT mirror (incl. padding)
    for (int e = tid; e < LCH * DVD; e += 256) {
      int t = e >> 7;
      Uf[e] = bet[t] * ((float)vc[e] - eG[t] * Uf[e]);
    }
    for (int e = tid; e < DVD * USTR; e += 256) uT[e] = (_Float16)0.f;
    __syncthreads();

    // ---- stage 4: blocked forward substitution (I+A)U = rhs ----
    for (int bi = 0; bi < 4; ++bi) {
      if (bi > 0) {  // cross-block: U[bi] += (-A)[bi][solved] * U[solved] via WMMA
        int tj = wave;
        v8f c = load_c(Uf, bi * 16, tj * 16, DVD);
#pragma unroll
        for (int kk = 0; kk < LCH; kk += 32) {
          v16h a = frag_a(anh, bi * 16, kk, LCH, 1);
          v16h bb = frag_b(uT, kk, tj * 16, 1, USTR);
          c = wmma16(a, bb, c);
        }
        store_c(Uf, bi * 16, tj * 16, DVD, c);
      }
      __syncthreads();
      if (tid < DVD) {  // 16x16 unit-lower-triangular diag block, per-column in regs
        int col = tid;
        float u[16];
#pragma unroll
        for (int r = 0; r < 16; ++r) {
          int row = bi * 16 + r;
          float val = Uf[row * DVD + col];
#pragma unroll
          for (int rp = 0; rp < 16; ++rp)
            if (rp < r) val += Ar[row * LCH + bi * 16 + rp] * u[rp];  // Ar = -A
          u[r] = val;
        }
#pragma unroll
        for (int r = 0; r < 16; ++r) {
          int row = bi * 16 + r;
          Uf[row * DVD + col] = u[r];
          uT[col * USTR + row] = (_Float16)u[r];  // transposed f16 mirror
        }
      }
      __syncthreads();
    }

    // ---- stage 5: O = Att*U + Qd*S0, streamed straight to global ----
    for (int job = wave; job < 32; job += 8) {
      int ti = job >> 3, tj = job & 7;
      v8f c = {};
#pragma unroll
      for (int kk = 0; kk < LCH; kk += 32) {
        v16h a = frag_a(atth, ti * 16, kk, LCH, 1);
        v16h bb = frag_b(uT, kk, tj * 16, 1, USTR);
        c = wmma16(a, bb, c);
      }
#pragma unroll
      for (int kk = 0; kk < DKD; kk += 32) {
        v16h a = frag_a(qc, ti * 16, kk, DKD, 1);
        v16h bb = frag_b(S16T, kk, tj * 16, 1, SSTR);
        c = wmma16(a, bb, c);
      }
      int lane = tid & 31;
      int nn = tj * 16 + (lane & 15);
      int mb = (lane >> 4) * 8;
#pragma unroll
      for (int i = 0; i < 8; ++i) {
        int t = ti * 16 + mb + i;
        size_t n = (size_t)(b * SS + tok0 + t);
        out[(n * HVH + hv) * DVD + nn] = c[i];
      }
    }
    __syncthreads();

    // ---- stage 6: S = expG_L * S + (decayed K)^T * U ----
    float eGL = eG[LCH - 1];
    for (int job = wave; job < 64; job += 8) {
      int ti = job >> 3, tj = job & 7;
      v8f c = load_c(S32, ti * 16, tj * 16, DVD);
#pragma unroll
      for (int i = 0; i < 8; ++i) c[i] *= eGL;
#pragma unroll
      for (int kk = 0; kk < LCH; kk += 32) {
        v16h a = frag_a(k2T, ti * 16, kk, K2STR, 1);   // [m=dk][k=t]: contiguous
        v16h bb = frag_b(uT, kk, tj * 16, 1, USTR);
        c = wmma16(a, bb, c);
      }
      int lane = tid & 31;
      int nn = tj * 16 + (lane & 15);
      int mb = (lane >> 4) * 8;
#pragma unroll
      for (int i = 0; i < 8; ++i) {
        int row = ti * 16 + mb + i;
        S32[row * DVD + nn] = c[i];
        S16T[nn * SSTR + row] = (_Float16)c[i];        // contiguous 16B runs per lane
      }
    }
    __syncthreads();
  }
}

static constexpr size_t SMEM_BYTES =
    (size_t)DKD * DVD * 4 +            // S32
    (size_t)LCH * DVD * 4 +            // Uf
    2u * LCH * LCH * 4 +               // Ar, QKr
    4u * LCH * 4 +                     // Gc, eG, w2, bet
    (size_t)DVD * SSTR * 2 +           // S16T
    3u * LCH * DKD * 2 +               // kc, qc, vc
    (size_t)DKD * K2STR * 2 +          // k2T
    (size_t)DVD * USTR * 2 +           // uT
    2u * LCH * LCH * 2;                // atth, anh   => 269,312 B (< 320KB WGP LDS)

extern "C" void kernel_launch(void* const* d_in, const int* in_sizes, int n_in,
                              void* d_out, int out_size, void* d_ws, size_t ws_size,
                              hipStream_t stream) {
  (void)in_sizes; (void)n_in; (void)out_size; (void)ws_size;
  const float* mixed = (const float*)d_in[0];
  const float* bvec  = (const float*)d_in[1];
  const float* avec  = (const float*)d_in[2];
  const float* cw    = (const float*)d_in[3];
  const float* dtb   = (const float*)d_in[4];
  const float* alog  = (const float*)d_in[5];
  // d_in[6] = cu_seqlens (equal-length batch; unused)

  const size_t NT = (size_t)BB * SS;  // 4096 tokens
  _Float16* qfp = (_Float16*)d_ws;
  _Float16* kfp = qfp + NT * HKH * DKD;
  _Float16* vfp = kfp + NT * HKH * DKD;
  float*    gbp = (float*)(vfp + NT * HVH * DVD);
  float*    bbp = gbp + NT * HVH;

  gdn_prep<<<(int)NT, 256, 0, stream>>>(mixed, bvec, avec, cw, dtb, alog,
                                        qfp, kfp, vfp, gbp, bbp);

  (void)hipFuncSetAttribute(reinterpret_cast<const void*>(gdn_chunks),
                            hipFuncAttributeMaxDynamicSharedMemorySize,
                            (int)SMEM_BYTES);
  gdn_chunks<<<BB * HVH, 256, SMEM_BYTES, stream>>>(qfp, kfp, vfp, gbp, bbp,
                                                    (float*)d_out);
}
